// ViT_24739011625767
// MI455X (gfx1250) — matrix operand order repaired and checked by reading, not compiled, not run on hardware
//
#include <hip/hip_runtime.h>
#include <math.h>
#include <stdint.h>

// ---------------------------------------------------------------------------
// ViT-Base forward (B=64, S=197, D=768, FF=3072, L=12, heads=8, dh=96)
// Compute-bound (~2.2 TFLOP vs <1 GB unique traffic @ 23.3 TB/s).
// Pipeline is f16-resident: weights converted/transposed to f16 N x K once
// per launch, activations produced in f16 by LN/attn/GELU. All GEMMs run
// v_wmma_f32_16x16x32_f16 with f32 accumulate. LDS tiles are fed by the
// Tensor Data Mover (tensor_load_to_lds, TENSORcnt), double-buffered.
// ---------------------------------------------------------------------------

typedef __attribute__((ext_vector_type(16))) _Float16 v16h;
typedef __attribute__((ext_vector_type(8)))  _Float16 v8h;
typedef __attribute__((ext_vector_type(8)))  float    v8f;
typedef __attribute__((ext_vector_type(4)))  uint32_t u32x4;
typedef __attribute__((ext_vector_type(8)))  uint32_t u32x8;

#define VB   64
#define VS   197
#define VD   768
#define VFF  3072
#define VNC  1000
#define VL   12
#define NHEADS 8
#define DHD  96
#define NPATCH 196          // 14x14
#define EPSLN 1e-5f

#define BM 128
#define BN 128
#define BK 32

// ---------------------------- reductions -----------------------------------
__device__ inline float block_sum(float v, float* red) {
    int tid = threadIdx.x;
    red[tid] = v; __syncthreads();
    for (int s = blockDim.x >> 1; s > 0; s >>= 1) {
        if (tid < s) red[tid] += red[tid + s];
        __syncthreads();
    }
    float r = red[0]; __syncthreads();
    return r;
}
__device__ inline float block_max(float v, float* red) {
    int tid = threadIdx.x;
    red[tid] = v; __syncthreads();
    for (int s = blockDim.x >> 1; s > 0; s >>= 1) {
        if (tid < s) red[tid] = fmaxf(red[tid], red[tid + s]);
        __syncthreads();
    }
    float r = red[0]; __syncthreads();
    return r;
}

// ---------------- TDM: load a 2D f16 tile (tileRows x BK) into LDS ----------
// D# per CDNA5 ISA. 2D tensor -> groups 2/3 NULL.
// rowsRem = remaining valid rows; TDM zero-fills OOB reads (handles M/N edge).
__device__ __forceinline__ void tdm_load_tile_f16(uint32_t ldsAddr,
                                                  const _Float16* gptr,
                                                  uint32_t rowsRem,
                                                  uint32_t tileRows,
                                                  uint32_t strideElems) {
    uint64_t ga = (uint64_t)(uintptr_t)gptr;
    u32x4 g0;
    g0.x = 1u;                                        // count=1 (valid), user mode
    g0.y = ldsAddr;                                   // LDS byte address
    g0.z = (uint32_t)ga;                              // global_addr[31:0]
    g0.w = ((uint32_t)(ga >> 32) & 0x01FFFFFFu)       // global_addr[56:32]
         | (2u << 30);                                // type = 2 ("image")
    u32x8 g1;
    g1[0] = 1u << 16;                                 // data_size = 1 -> 2 bytes
    g1[1] = ((uint32_t)BK & 0xFFFFu) << 16;           // tensor_dim0[15:0] = 32
    g1[2] = (rowsRem & 0xFFFFu) << 16;                // dim0[31:16]=0 | tensor_dim1 lo
    g1[3] = ((rowsRem >> 16) & 0xFFFFu)               // tensor_dim1 hi
          | ((uint32_t)BK << 16);                     // tile_dim0 = 32
    g1[4] = tileRows & 0xFFFFu;                       // tile_dim1 | tile_dim2 = 0
    g1[5] = strideElems;                              // tensor_dim0_stride[31:0]
    g1[6] = 0u;                                       // stride hi | dim1_stride lo
    g1[7] = 0u;
    asm volatile("tensor_load_to_lds %0, %1" :: "s"(g0), "s"(g1) : "memory");
}

// ------------------- per-sample mean/rstd over `count` ---------------------
__global__ __launch_bounds__(512) void sample_stats(const float* __restrict__ src,
                                                    long long count,
                                                    float* __restrict__ stats) {
    __shared__ float red[512];
    int b = blockIdx.x;
    const float* p = src + (size_t)b * count;
    float s = 0.f, ss = 0.f;
    for (long long i = threadIdx.x; i < count; i += blockDim.x) {
        float v = p[i]; s += v; ss += v * v;
    }
    s  = block_sum(s, red);
    ss = block_sum(ss, red);
    if (threadIdx.x == 0) {
        float m = s / (float)count;
        float var = ss / (float)count - m * m;
        stats[b * 2 + 0] = m;
        stats[b * 2 + 1] = rsqrtf(var + EPSLN);
    }
}

// ------------------------ weight fp32 -> f16 helpers ------------------------
__global__ __launch_bounds__(256) void convert_f16(const float* __restrict__ in,
                                                   _Float16* __restrict__ out,
                                                   long long n) {
    long long i = (long long)blockIdx.x * blockDim.x + threadIdx.x;
    if (i < n) out[i] = (_Float16)in[i];
}
// per-layer K x N fp32 -> N x K f16 (blockIdx.y = layer)
__global__ __launch_bounds__(256) void transpose_f16(const float* __restrict__ in,
                                                     _Float16* __restrict__ out,
                                                     int K, int N) {
    int l = blockIdx.y;
    const float* src = in + (size_t)l * K * N;
    _Float16* dst = out + (size_t)l * K * N;
    long long i = (long long)blockIdx.x * blockDim.x + threadIdx.x;
    if (i < (long long)K * N) {
        int k = (int)(i / N), n2 = (int)(i % N);
        dst[(size_t)n2 * K + k] = (_Float16)src[i];
    }
}

// ------- im2col of LN0-normalized input: A[row=B*196][k=c*256+py*16+px] -----
__global__ __launch_bounds__(256) void im2col_ln0(const float* __restrict__ x,
                                                  const float* __restrict__ w,
                                                  const float* __restrict__ bco,
                                                  const float* __restrict__ stats,
                                                  _Float16* __restrict__ Aout) {
    int row = blockIdx.x;             // 0..12543
    int b = row / NPATCH, p = row % NPATCH;
    int py0 = (p / 14) * 16, px0 = (p % 14) * 16;
    float m = stats[b * 2 + 0], r = stats[b * 2 + 1];
    for (int col = threadIdx.x; col < VD; col += blockDim.x) {
        int c = col >> 8, rem = col & 255;
        int dy = rem >> 4, dx = rem & 15;
        int gy = py0 + dy, gx = px0 + dx;
        size_t xi = (((size_t)b * 3 + c) * 224 + gy) * 224 + gx;
        int wi = (c * 224 + gy) * 224 + gx;
        Aout[(size_t)row * VD + col] = (_Float16)((x[xi] - m) * r * w[wi] + bco[wi]);
    }
}

// -------- h[b,s,:] = (s==0 ? cls : LN1(patch)) + pos_emb[s,:] --------------
__global__ __launch_bounds__(256) void build_h(const float* __restrict__ y,
                                               const float* __restrict__ ln1w,
                                               const float* __restrict__ ln1b,
                                               const float* __restrict__ stats,
                                               const float* __restrict__ cls,
                                               const float* __restrict__ pos,
                                               float* __restrict__ h) {
    int idx = blockIdx.x;             // 0..B*S-1
    int b = idx / VS, s = idx % VS;
    float* out = h + ((size_t)b * VS + s) * VD;
    if (s == 0) {
        for (int d = threadIdx.x; d < VD; d += blockDim.x)
            out[d] = cls[d] + pos[d];
    } else {
        int p = s - 1;
        float m = stats[b * 2 + 0], r = stats[b * 2 + 1];
        const float* src = y + ((size_t)b * NPATCH + p) * VD;
        for (int d = threadIdx.x; d < VD; d += blockDim.x) {
            int wi = d * NPATCH + p;   // ln1_w is (D, 14, 14)
            out[d] = (src[d] - m) * r * ln1w[wi] + ln1b[wi] + pos[s * VD + d];
        }
    }
}

// ------------------- per-row LayerNorm, fp32 in -> f16 out ------------------
__global__ __launch_bounds__(256) void ln_rows(const float* __restrict__ src,
                                               long long srcStride,
                                               const float* __restrict__ w,
                                               const float* __restrict__ bvec,
                                               _Float16* __restrict__ dst,
                                               long long dstStride, int n) {
    __shared__ float red[256];
    int row = blockIdx.x;
    const float* p = src + (size_t)row * srcStride;
    float s = 0.f, ss = 0.f;
    for (int i = threadIdx.x; i < n; i += blockDim.x) {
        float v = p[i]; s += v; ss += v * v;
    }
    s  = block_sum(s, red);
    ss = block_sum(ss, red);
    float m = s / (float)n;
    float inv = rsqrtf(ss / (float)n - m * m + EPSLN);
    _Float16* o = dst + (size_t)row * dstStride;
    for (int i = threadIdx.x; i < n; i += blockDim.x)
        o[i] = (_Float16)((p[i] - m) * inv * w[i] + bvec[i]);
}

// ------------------------------ WMMA GEMM -----------------------------------
// C[M,N] = A[M,K](f16) * Bt[N,K](f16, pre-transposed) + bias (+residual)
// (gelu?). Output to Cf (fp32) or Ch (f16). LDS tiles fed by TDM, double-buf.
__global__ __launch_bounds__(256) void gemm_wmma(const _Float16* __restrict__ A,
                                                 const _Float16* __restrict__ Bt,
                                                 const float* __restrict__ bias,
                                                 const float* __restrict__ residual,
                                                 float* __restrict__ Cf,
                                                 _Float16* __restrict__ Ch,
                                                 int M, int N, int K, int gelu) {
    __shared__ _Float16 sA[2][BM * BK];    // row-major [m][k]
    __shared__ _Float16 sB[2][BN * BK];    // [n][k] (Bt rows are contiguous)

    int tid  = threadIdx.x;
    int lane = tid & 31;
    int wave = tid >> 5;                   // 8 waves
    int waveM = wave >> 2;                 // 0..1 -> 64 rows each
    int waveN = wave & 3;                  // 0..3 -> 32 cols each
    int tileM = blockIdx.y * BM;
    int tileN = blockIdx.x * BN;

    v8f acc[4][2];
#pragma unroll
    for (int i = 0; i < 4; ++i)
#pragma unroll
        for (int j = 0; j < 2; ++j) acc[i][j] = (v8f)0.f;

    const int mfrag = lane & 15;
    const int kA = (lane >> 4) * 8;        // A frag: K-halves at kA.. and kA+16..
    const int kB = (lane >> 4) * 8;        // B frag: K-halves at kB.. and kB+16..

    const uint32_t rowsA = (uint32_t)(M - tileM);
    const uint32_t rowsB = (uint32_t)(N - tileN);

    // prologue: stage k0 = 0 into buffer 0 (wave 0 only; TENSORcnt is per-wave)
    if (tid < 32) {
        tdm_load_tile_f16((uint32_t)(uintptr_t)&sA[0][0],
                          A + (size_t)tileM * K, rowsA, BM, (uint32_t)K);
        tdm_load_tile_f16((uint32_t)(uintptr_t)&sB[0][0],
                          Bt + (size_t)tileN * K, rowsB, BN, (uint32_t)K);
    }

    const int nsteps = K / BK;
    for (int step = 0; step < nsteps; ++step) {
        const int cur = step & 1;
        const int kNext = (step + 1) * BK;
        if (tid < 32) {
            if (kNext < K) {
                tdm_load_tile_f16((uint32_t)(uintptr_t)&sA[cur ^ 1][0],
                                  A + (size_t)tileM * K + kNext, rowsA, BM, (uint32_t)K);
                tdm_load_tile_f16((uint32_t)(uintptr_t)&sB[cur ^ 1][0],
                                  Bt + (size_t)tileN * K + kNext, rowsB, BN, (uint32_t)K);
                // TDM completes in order: <=2 outstanding means current stage done
                __builtin_amdgcn_s_wait_tensorcnt(2);
            } else {
                __builtin_amdgcn_s_wait_tensorcnt(0);
            }
        }
        __syncthreads();                   // cur buffers visible to all waves

        v16h bfrag[2];
#pragma unroll
        for (int j = 0; j < 2; ++j) {
            int n = waveN * 32 + j * 16 + mfrag;
            v8h lo = *(const v8h*)(&sB[cur][n * BK + kB]);
            v8h hi = *(const v8h*)(&sB[cur][n * BK + kB + 16]);
#pragma unroll
            for (int t = 0; t < 8; ++t) { bfrag[j][t] = lo[t]; bfrag[j][t + 8] = hi[t]; }
        }
#pragma unroll
        for (int i = 0; i < 4; ++i) {
            int m = waveM * 64 + i * 16 + mfrag;
            v8h lo = *(const v8h*)(&sA[cur][m * BK + kA]);
            v8h hi = *(const v8h*)(&sA[cur][m * BK + kA + 16]);
            v16h af;
#pragma unroll
            for (int t = 0; t < 8; ++t) { af[t] = lo[t]; af[t + 8] = hi[t]; }
#pragma unroll
            for (int j = 0; j < 2; ++j) {
                acc[i][j] = __builtin_amdgcn_wmma_f32_16x16x32_f16(
                    false, af, false, bfrag[j], (short)0, acc[i][j], false, false);
            }
        }
        __syncthreads();                   // done with cur before TDM reuses it
    }

    // ---- store: D layout: VGPR r -> row r + 8*(lane>=16); col = lane&15 ----
    int rBase = (lane >> 4) * 8;
    int cl = lane & 15;
#pragma unroll
    for (int i = 0; i < 4; ++i) {
#pragma unroll
        for (int j = 0; j < 2; ++j) {
            int col = tileN + waveN * 32 + j * 16 + cl;
#pragma unroll
            for (int r = 0; r < 8; ++r) {
                int row = tileM + waveM * 64 + i * 16 + rBase + r;
                if (row < M && col < N) {
                    float v = acc[i][j][r];
                    if (bias)     v += bias[col];
                    if (residual) v += residual[(size_t)row * N + col];
                    if (gelu)     v = 0.5f * v * (1.f + erff(v * 0.70710678118654752f));
                    if (Cf) Cf[(size_t)row * N + col] = v;
                    else    Ch[(size_t)row * N + col] = (_Float16)v;
                }
            }
        }
    }
}

// ------------------------- attention (per b,h,q) ----------------------------
__global__ __launch_bounds__(256) void attn_kernel(const _Float16* __restrict__ Q,
                                                   const _Float16* __restrict__ Km,
                                                   const _Float16* __restrict__ V,
                                                   _Float16* __restrict__ O,
                                                   float invscale) {
    __shared__ float qv[DHD];
    __shared__ float sc[VS];
    __shared__ float red[256];
    int qs = blockIdx.x, hh = blockIdx.y, b = blockIdx.z;
    int tid = threadIdx.x;
    const _Float16* qrow = Q + ((size_t)b * VS + qs) * VD + hh * DHD;
    if (tid < DHD) qv[tid] = (float)qrow[tid];
    __syncthreads();

    float lmax = -1e30f;
    for (int k = tid; k < VS; k += blockDim.x) {
        const _Float16* krow = Km + ((size_t)b * VS + k) * VD + hh * DHD;
        float d = 0.f;
#pragma unroll 8
        for (int i = 0; i < DHD; ++i) d += qv[i] * (float)krow[i];
        d *= invscale;
        sc[k] = d;
        lmax = fmaxf(lmax, d);
    }
    float m = block_max(lmax, red);
    float lsum = 0.f;
    for (int k = tid; k < VS; k += blockDim.x) {
        float e = __expf(sc[k] - m);
        sc[k] = e;
        lsum += e;
    }
    float ssum = block_sum(lsum, red);   // syncs -> sc fully written
    float inv = 1.f / ssum;
    for (int d = tid; d < DHD; d += blockDim.x) {
        float a = 0.f;
        for (int k = 0; k < VS; ++k)
            a += sc[k] * (float)V[((size_t)b * VS + k) * VD + hh * DHD + d];
        O[((size_t)b * VS + qs) * VD + hh * DHD + d] = (_Float16)(a * inv);
    }
}

// ---------------------------- final softmax ---------------------------------
__global__ __launch_bounds__(256) void softmax_rows(const float* __restrict__ logits,
                                                    float* __restrict__ out, int n) {
    __shared__ float red[256];
    int row = blockIdx.x;
    const float* p = logits + (size_t)row * n;
    float lm = -1e30f;
    for (int i = threadIdx.x; i < n; i += blockDim.x) lm = fmaxf(lm, p[i]);
    float m = block_max(lm, red);
    float ls = 0.f;
    for (int i = threadIdx.x; i < n; i += blockDim.x) ls += __expf(p[i] - m);
    float s = block_sum(ls, red);
    float inv = 1.f / s;
    for (int i = threadIdx.x; i < n; i += blockDim.x)
        out[(size_t)row * n + i] = __expf(p[i] - m) * inv;
}

// ----------------------------------------------------------------------------
extern "C" void kernel_launch(void* const* d_in, const int* in_sizes, int n_in,
                              void* d_out, int out_size, void* d_ws, size_t ws_size,
                              hipStream_t stream) {
    (void)in_sizes; (void)n_in; (void)out_size; (void)ws_size;
    const float* x      = (const float*)d_in[0];
    const float* ln0_w  = (const float*)d_in[1];
    const float* ln0_b  = (const float*)d_in[2];
    const float* conv_w = (const float*)d_in[3];   // (D, C*P*P) = N x K already
    const float* conv_b = (const float*)d_in[4];
    const float* ln1_w  = (const float*)d_in[5];
    const float* ln1_b  = (const float*)d_in[6];
    const float* cls_t  = (const float*)d_in[7];
    const float* pos    = (const float*)d_in[8];
    const float* lnA_w  = (const float*)d_in[9];
    const float* lnA_b  = (const float*)d_in[10];
    const float* Wq     = (const float*)d_in[11];
    const float* bq     = (const float*)d_in[12];
    const float* Wk     = (const float*)d_in[13];
    const float* bk     = (const float*)d_in[14];
    const float* Wv     = (const float*)d_in[15];
    const float* bv     = (const float*)d_in[16];
    const float* Wo     = (const float*)d_in[17];
    const float* bo     = (const float*)d_in[18];
    const float* lnB_w  = (const float*)d_in[19];
    const float* lnB_b  = (const float*)d_in[20];
    const float* W1     = (const float*)d_in[21];
    const float* b1     = (const float*)d_in[22];
    const float* W2     = (const float*)d_in[23];
    const float* b2     = (const float*)d_in[24];
    const float* hln_w  = (const float*)d_in[25];
    const float* hln_b  = (const float*)d_in[26];
    const float* hW     = (const float*)d_in[27];  // (D, NC) = K x N
    const float* hb     = (const float*)d_in[28];

    const int Mtok = VB * VS;          // 12608
    const int Mpat = VB * NPATCH;      // 12544
    const size_t tokD = (size_t)Mtok * VD;

    // ---------------- workspace layout (256B-aligned regions) ---------------
    char* wp = (char*)d_ws;
    auto alloc = [&](size_t bytes) -> void* {
        void* r = (void*)wp;
        wp += (bytes + 255) & ~(size_t)255;
        return r;
    };
    float*    stats0 = (float*)alloc(VB * 2 * sizeof(float));
    float*    stats1 = (float*)alloc(VB * 2 * sizeof(float));
    float*    logits = (float*)alloc((size_t)VB * VNC * sizeof(float));
    float*    h      = (float*)alloc(tokD * sizeof(float));
    _Float16* y      = (_Float16*)alloc(tokD * sizeof(_Float16));
    _Float16* q      = (_Float16*)alloc(tokD * sizeof(_Float16));
    _Float16* k      = (_Float16*)alloc(tokD * sizeof(_Float16));
    _Float16* v      = (_Float16*)alloc(tokD * sizeof(_Float16));
    _Float16* o      = (_Float16*)alloc(tokD * sizeof(_Float16));
    _Float16* ff1    = (_Float16*)alloc((size_t)Mtok * VFF * sizeof(_Float16));
    // patch-embed scratch aliases ff1 (only live before the layer loop):
    _Float16* Acol   = ff1;                                   // 12544 x 768 f16
    float*    y0     = (float*)((char*)ff1 + (size_t)Mpat * VD * sizeof(_Float16));
    _Float16* clsbuf = (_Float16*)alloc((size_t)VB * VD * sizeof(_Float16));
    _Float16* convT  = (_Float16*)alloc((size_t)VD * VD * sizeof(_Float16));
    _Float16* WqT    = (_Float16*)alloc((size_t)VL * VD * VD * sizeof(_Float16));
    _Float16* WkT    = (_Float16*)alloc((size_t)VL * VD * VD * sizeof(_Float16));
    _Float16* WvT    = (_Float16*)alloc((size_t)VL * VD * VD * sizeof(_Float16));
    _Float16* WoT    = (_Float16*)alloc((size_t)VL * VD * VD * sizeof(_Float16));
    _Float16* W1T    = (_Float16*)alloc((size_t)VL * VD * VFF * sizeof(_Float16));
    _Float16* W2T    = (_Float16*)alloc((size_t)VL * VFF * VD * sizeof(_Float16));
    _Float16* hWT    = (_Float16*)alloc((size_t)VNC * VD * sizeof(_Float16));

    dim3 blk(256);
    auto gemmGrid = [](int M, int N) { return dim3((N + BN - 1) / BN, (M + BM - 1) / BM); };
    const float invscale = rsqrtf((float)VD);

    // ---- weight prep: fp32 -> f16, transposed to N x K (TDM-friendly) ----
    {
        long long ndd = (long long)VD * VD, ndf = (long long)VD * VFF;
        dim3 gdd((unsigned)((ndd + 255) / 256), VL), gdf((unsigned)((ndf + 255) / 256), VL);
        transpose_f16<<<gdd, blk, 0, stream>>>(Wq, WqT, VD, VD);
        transpose_f16<<<gdd, blk, 0, stream>>>(Wk, WkT, VD, VD);
        transpose_f16<<<gdd, blk, 0, stream>>>(Wv, WvT, VD, VD);
        transpose_f16<<<gdd, blk, 0, stream>>>(Wo, WoT, VD, VD);
        transpose_f16<<<gdf, blk, 0, stream>>>(W1, W1T, VD, VFF);
        transpose_f16<<<gdf, blk, 0, stream>>>(W2, W2T, VFF, VD);
        long long nh = (long long)VD * VNC;
        transpose_f16<<<dim3((unsigned)((nh + 255) / 256), 1), blk, 0, stream>>>(hW, hWT, VD, VNC);
        convert_f16<<<(unsigned)((ndd + 255) / 256), blk, 0, stream>>>(conv_w, convT, ndd);
    }

    // ---- patch embedding ----
    sample_stats<<<VB, 512, 0, stream>>>(x, 3LL * 224 * 224, stats0);
    im2col_ln0<<<Mpat, blk, 0, stream>>>(x, ln0_w, ln0_b, stats0, Acol);
    gemm_wmma<<<gemmGrid(Mpat, VD), blk, 0, stream>>>(Acol, convT, conv_b, nullptr,
                                                      y0, nullptr, Mpat, VD, VD, 0);
    sample_stats<<<VB, 512, 0, stream>>>(y0, (long long)NPATCH * VD, stats1);
    build_h<<<Mtok, blk, 0, stream>>>(y0, ln1_w, ln1_b, stats1, cls_t, pos, h);

    // ---- transformer layers ----
    for (int l = 0; l < VL; ++l) {
        const size_t wOff = (size_t)l * VD * VD;
        ln_rows<<<Mtok, blk, 0, stream>>>(h, VD, lnA_w + l * VD, lnA_b + l * VD, y, VD, VD);
        gemm_wmma<<<gemmGrid(Mtok, VD), blk, 0, stream>>>(y, WqT + wOff, bq + l * VD, nullptr,
                                                          nullptr, q, Mtok, VD, VD, 0);
        gemm_wmma<<<gemmGrid(Mtok, VD), blk, 0, stream>>>(y, WkT + wOff, bk + l * VD, nullptr,
                                                          nullptr, k, Mtok, VD, VD, 0);
        gemm_wmma<<<gemmGrid(Mtok, VD), blk, 0, stream>>>(y, WvT + wOff, bv + l * VD, nullptr,
                                                          nullptr, v, Mtok, VD, VD, 0);
        attn_kernel<<<dim3(VS, NHEADS, VB), blk, 0, stream>>>(q, k, v, o, invscale);
        gemm_wmma<<<gemmGrid(Mtok, VD), blk, 0, stream>>>(o, WoT + wOff, bo + l * VD, h,
                                                          h, nullptr, Mtok, VD, VD, 0);
        ln_rows<<<Mtok, blk, 0, stream>>>(h, VD, lnB_w + l * VD, lnB_b + l * VD, y, VD, VD);
        gemm_wmma<<<gemmGrid(Mtok, VFF), blk, 0, stream>>>(y, W1T + (size_t)l * VD * VFF,
                                                           b1 + l * VFF, nullptr,
                                                           nullptr, ff1, Mtok, VFF, VD, 1);
        gemm_wmma<<<gemmGrid(Mtok, VD), blk, 0, stream>>>(ff1, W2T + (size_t)l * VFF * VD,
                                                          b2 + l * VD, h,
                                                          h, nullptr, Mtok, VD, VFF, 0);
    }

    // ---- head ----
    ln_rows<<<VB, blk, 0, stream>>>(h, (long long)VS * VD, hln_w, hln_b, clsbuf, VD, VD);
    gemm_wmma<<<gemmGrid(VB, VNC), blk, 0, stream>>>(clsbuf, hWT, hb, nullptr,
                                                     logits, nullptr, VB, VNC, VD, 0);
    softmax_rows<<<VB, blk, 0, stream>>>(logits, (float*)d_out, VNC);
}
